// RULSTM_60997125538359
// MI455X (gfx1250) — matrix operand order repaired and checked by reading, not compiled, not run on hardware
//
#include <hip/hip_runtime.h>
#include <hip/hip_bf16.h>
#include <math.h>

typedef __bf16 bf16;
typedef __attribute__((ext_vector_type(16))) __bf16 v16bf;
typedef __attribute__((ext_vector_type(8)))  __bf16 v8bf;
typedef __attribute__((ext_vector_type(8)))  float  v8f;

#define S_    16
#define B_    128
#define FEAT_ 2048
#define HAND_ 63
#define H_    1024
#define C_    1000
#define CPAD_ 1024

__device__ __forceinline__ float sigmoidf_(float x)
{
    return 1.f / (1.f + __expf(-x));
}

// ---- split register-relay staging of one 16x128 bf16 A chunk -------------
// load_chunk issues the global_load early; store_chunk does the LDS store
// late (after the compute burst) so the load latency hides under WMMAs.
__device__ __forceinline__ v8bf load_chunk(const bf16* __restrict__ A, int lda,
                                           int row0, int kc)
{
    const int r  = threadIdx.x >> 4;    // 0..15 row
    const int ch = threadIdx.x & 15;    // 0..15 sixteen-byte chunk
    return *(const v8bf*)(A + (size_t)(row0 + r) * lda + kc + ch * 8);
}
__device__ __forceinline__ void store_chunk(bf16* dst, v8bf v)
{
    const int r  = threadIdx.x >> 4;
    const int ch = threadIdx.x & 15;
    *(v8bf*)(dst + r * 128 + ch * 8) = v;
}

// ---------------------------------------------------------------------------
// WMMA GEMM:  C[M x N] = A[M x K](bf16) @ W[N x K](bf16)^T (+bias[N])
// M = gridDim.y*16.  N covered = gridDim.x * 8 * NTILES*16; stores masked to n < nmax.
// Double-buffered LDS A staging with deferred ds_store; all NTILES B
// fragments preloaded before the WMMA burst.
// ---------------------------------------------------------------------------
template<int NTILES>
__global__ __launch_bounds__(256)
void wmma_gemm(const bf16* __restrict__ A, int lda,
               const bf16* __restrict__ W, int ldw,
               int K,
               const float* __restrict__ bias,
               float* __restrict__ Cf, int ldc,
               bf16*  __restrict__ Cb, int ldcb,
               int nmax)
{
    constexpr int KC = 128;
    __shared__ bf16 sA[2][16 * KC];
    const int lane  = threadIdx.x & 31;
    const int wave  = threadIdx.x >> 5;
    const int row0  = blockIdx.y * 16;
    const int lo    = lane & 15;
    const int hi    = lane >> 4;
    const int nbase = (blockIdx.x * 8 + wave) * (NTILES * 16);

    v8f acc[NTILES];
    #pragma unroll
    for (int t = 0; t < NTILES; ++t)
        #pragma unroll
        for (int i = 0; i < 8; ++i) acc[t][i] = 0.f;

    store_chunk(sA[0], load_chunk(A, lda, row0, 0));
    __syncthreads();

    for (int kc = 0; kc < K; kc += KC) {
        const int cur  = (kc / KC) & 1;
        const bool more = (kc + KC < K);
        v8bf pend;
        if (more) pend = load_chunk(A, lda, row0, kc + KC);  // issue early

        #pragma unroll
        for (int kk = 0; kk < KC; kk += 32) {
            // preload ALL B fragments first (loads overlap WMMAs)
            v16bf bv[NTILES];
            #pragma unroll
            for (int t = 0; t < NTILES; ++t) {
                const bf16* wp = W + (size_t)(nbase + t * 16 + lo) * ldw
                                   + kc + kk + hi * 16;
                v8bf b0 = *(const v8bf*)(wp);
                v8bf b1 = *(const v8bf*)(wp + 8);
                #pragma unroll
                for (int i = 0; i < 8; ++i) { bv[t][i] = b0[i]; bv[t][i + 8] = b1[i]; }
            }
            // A fragment (contiguous k = 16*hi + s, matching B)
            const bf16* ap = &sA[cur][lo * KC + kk + hi * 16];
            v8bf a0 = *(const v8bf*)(ap);
            v8bf a1 = *(const v8bf*)(ap + 8);
            v16bf av;
            #pragma unroll
            for (int i = 0; i < 8; ++i) { av[i] = a0[i]; av[i + 8] = a1[i]; }

            #pragma unroll
            for (int t = 0; t < NTILES; ++t)
                acc[t] = __builtin_amdgcn_wmma_f32_16x16x32_bf16(
                    false, av, false, bv[t], (short)0, acc[t], false, false);
        }

        if (more) store_chunk(sA[cur ^ 1], pend);            // store late
        __syncthreads();
    }

    #pragma unroll
    for (int t = 0; t < NTILES; ++t) {
        const int n = nbase + t * 16 + lo;
        const bool nvalid = (n < nmax);
        const float bval = (bias != nullptr && nvalid) ? bias[n] : 0.f;
        #pragma unroll
        for (int g = 0; g < 8; ++g) {
            const int m = row0 + g + hi * 8;
            float v = acc[t][g] + bval;
            if (nvalid) {
                if (Cf) Cf[(size_t)m * ldc + n] = v;
                else    Cb[(size_t)m * ldcb + n] = (bf16)v;
            }
        }
    }
}

// ---------------------------------------------------------------------------
// Fused LSTM step:  gates = h_in @ Whh^T + Gt  (gate rows permuted p=4*j+g),
// then the LSTM nonlinearity in the epilogue via a 16KB LDS gate buffer.
// Block = 16 batch rows x 256 gate cols = all 4 gates of 64 channels.
// grid = (4H/256, B/16).  h double-buffered across launches; c in-place.
// ---------------------------------------------------------------------------
__global__ __launch_bounds__(256)
void lstm_step_fused(const bf16* __restrict__ hin,   // B x H
                     const bf16* __restrict__ Whh,   // 4H x H, rows p=4j+g
                     const float* __restrict__ Gt,   // B x 4H, permuted
                     float* __restrict__ c,          // B x H (in/out)
                     bf16*  __restrict__ hout)       // B x H
{
    constexpr int KC = 128, NTILES = 2, K = H_;
    __shared__ __align__(16) char smem[16 * 256 * sizeof(float)];  // 16 KB
    bf16*  sAb = (bf16*)smem;          // 2 x (16x128) staging = 8 KB
    float* sG  = (float*)smem;         // 16 x 256 gates (after last barrier)

    const int lane  = threadIdx.x & 31;
    const int wave  = threadIdx.x >> 5;
    const int row0  = blockIdx.y * 16;
    const int lo    = lane & 15;
    const int hi    = lane >> 4;
    const int nbase = (blockIdx.x * 8 + wave) * (NTILES * 16);

    v8f acc[NTILES];
    #pragma unroll
    for (int t = 0; t < NTILES; ++t)
        #pragma unroll
        for (int i = 0; i < 8; ++i) acc[t][i] = 0.f;

    store_chunk(sAb, load_chunk(hin, H_, row0, 0));
    __syncthreads();

    for (int kc = 0; kc < K; kc += KC) {
        const int cur  = (kc / KC) & 1;
        const bool more = (kc + KC < K);
        v8bf pend;
        if (more) pend = load_chunk(hin, H_, row0, kc + KC);

        #pragma unroll
        for (int kk = 0; kk < KC; kk += 32) {
            v16bf bv[NTILES];
            #pragma unroll
            for (int t = 0; t < NTILES; ++t) {
                const bf16* wp = Whh + (size_t)(nbase + t * 16 + lo) * H_
                                     + kc + kk + hi * 16;
                v8bf b0 = *(const v8bf*)(wp);
                v8bf b1 = *(const v8bf*)(wp + 8);
                #pragma unroll
                for (int i = 0; i < 8; ++i) { bv[t][i] = b0[i]; bv[t][i + 8] = b1[i]; }
            }
            const bf16* ap = sAb + cur * 16 * KC + lo * KC + kk + hi * 16;
            v8bf a0 = *(const v8bf*)(ap);
            v8bf a1 = *(const v8bf*)(ap + 8);
            v16bf av;
            #pragma unroll
            for (int i = 0; i < 8; ++i) { av[i] = a0[i]; av[i + 8] = a1[i]; }

            #pragma unroll
            for (int t = 0; t < NTILES; ++t)
                acc[t] = __builtin_amdgcn_wmma_f32_16x16x32_bf16(
                    false, av, false, bv[t], (short)0, acc[t], false, false);
        }

        if (more) store_chunk(sAb + (cur ^ 1) * 16 * KC, pend);
        __syncthreads();   // after last iter, sA is dead -> sG may alias it
    }

    // ---- epilogue 1: gates + addend -> LDS ----
    #pragma unroll
    for (int t = 0; t < NTILES; ++t) {
        const int nl = wave * 32 + t * 16 + lo;          // 0..255 local col
        const int n  = blockIdx.x * 256 + nl;            // global gate col
        #pragma unroll
        for (int g = 0; g < 8; ++g) {
            const int r = g + 8 * hi;                    // 0..15 local row
            sG[r * 256 + nl] = acc[t][g] + Gt[(size_t)(row0 + r) * (4 * H_) + n];
        }
    }
    __syncthreads();

    // ---- epilogue 2: LSTM nonlinearity on 16 rows x 64 channels ----
    const int jbase = blockIdx.x * 64;
    #pragma unroll
    for (int it = 0; it < 4; ++it) {
        const int q  = it * 256 + threadIdx.x;           // 0..1023
        const int r  = q >> 6;                           // 0..15
        const int jl = q & 63;                           // 0..63
        const float4 gv = *(const float4*)(&sG[r * 256 + jl * 4]); // i,f,g,o
        const size_t idx = (size_t)(row0 + r) * H_ + (jbase + jl);
        float cn = sigmoidf_(gv.y) * c[idx] + sigmoidf_(gv.x) * tanhf(gv.z);
        c[idx]   = cn;
        hout[idx] = (bf16)(sigmoidf_(gv.w) * tanhf(cn));
    }
}

// ---------------------------------------------------------------------------
// Small helper kernels
// ---------------------------------------------------------------------------
__global__ void cvt_f32_bf16(const float* __restrict__ src,
                             bf16* __restrict__ dst, int n)
{
    int i = blockIdx.x * blockDim.x + threadIdx.x;
    if (i < n) dst[i] = (bf16)src[i];
}

// gate weight (4H x Kdim): dst row p=4*j+g  <-  src row g*H + j
__global__ void cvt_w_perm(const float* __restrict__ src,
                           bf16* __restrict__ dst, int Kdim)
{
    int i = blockIdx.x * blockDim.x + threadIdx.x;
    if (i >= 4 * H_ * Kdim) return;
    int p = i / Kdim, k = i - p * Kdim;
    int j = p >> 2, g = p & 3;
    dst[i] = (bf16)src[(size_t)(g * H_ + j) * Kdim + k];
}

__global__ void bias_sum_perm(const float* __restrict__ a,
                              const float* __restrict__ b,
                              float* __restrict__ dst)
{
    int p = blockIdx.x * blockDim.x + threadIdx.x;
    if (p >= 4 * H_) return;
    int s = (p & 3) * H_ + (p >> 2);
    dst[p] = a[s] + b[s];
}

// fch_w (1024 x 126) -> (1024 x 128) zero-padded
__global__ void cvt_fch(const float* __restrict__ src, bf16* __restrict__ dst)
{
    int i = blockIdx.x * blockDim.x + threadIdx.x;
    if (i >= H_ * 128) return;
    int r = i >> 7, cc = i & 127;
    dst[i] = (cc < 2 * HAND_) ? (bf16)src[r * (2 * HAND_) + cc] : (bf16)0.f;
}

// cls_w (1000 x 1024) -> (1024 x 1024) zero row padding
__global__ void cvt_cls(const float* __restrict__ src, bf16* __restrict__ dst)
{
    int i = blockIdx.x * blockDim.x + threadIdx.x;
    if (i >= CPAD_ * H_) return;
    dst[i] = ((i / H_) < C_) ? (bf16)src[i] : (bf16)0.f;
}

// hands = [lh, rh] -> (S*B x 128) bf16, cols 126/127 zero
__global__ void pack_hands(const float* __restrict__ lh,
                           const float* __restrict__ rh,
                           bf16* __restrict__ dst)
{
    int i = blockIdx.x * blockDim.x + threadIdx.x;
    if (i >= S_ * B_ * 128) return;
    int t = i >> 7, cc = i & 127;
    float v = 0.f;
    if (cc < HAND_)           v = lh[t * HAND_ + cc];
    else if (cc < 2 * HAND_)  v = rh[t * HAND_ + (cc - HAND_)];
    dst[i] = (bf16)v;
}

__global__ void zero_hc(float* __restrict__ c, bf16* __restrict__ h)
{
    int i = blockIdx.x * blockDim.x + threadIdx.x;
    if (i < B_ * H_) { c[i] = 0.f; h[i] = (bf16)0.f; }
}

// argmax + logsumexp loss; one block, one row per thread
__global__ __launch_bounds__(128)
void head_kernel(const float* __restrict__ out, const int* __restrict__ act,
                 float* __restrict__ pred, float* __restrict__ loss)
{
    __shared__ float red[128];
    const int b = threadIdx.x;
    const float* row = out + (size_t)b * C_;
    float m = row[0]; int am = 0;
    for (int j = 1; j < C_; ++j) {
        float v = row[j];
        if (v > m) { m = v; am = j; }
    }
    float s = 0.f;
    for (int j = 0; j < C_; ++j) s += __expf(row[j] - m);
    float logz = m + __logf(s);
    pred[b] = (float)am;
    red[b] = logz - row[act[b]];
    __syncthreads();
    for (int off = 64; off > 0; off >>= 1) {
        if (b < off) red[b] += red[b + off];
        __syncthreads();
    }
    if (b == 0) loss[0] = red[0] / (float)B_;
}

// ---------------------------------------------------------------------------
// Orchestration
// ---------------------------------------------------------------------------
extern "C" void kernel_launch(void* const* d_in, const int* in_sizes, int n_in,
                              void* d_out, int out_size, void* d_ws, size_t ws_size,
                              hipStream_t stream)
{
    (void)in_sizes; (void)n_in; (void)out_size; (void)ws_size;

    const float* feats      = (const float*)d_in[0];
    const float* lh         = (const float*)d_in[1];
    const float* rh         = (const float*)d_in[2];
    const int*   act        = (const int*)  d_in[3];
    const float* fc1_w      = (const float*)d_in[4];
    const float* fc1_b      = (const float*)d_in[5];
    const float* fch_w      = (const float*)d_in[6];
    const float* fch_b      = (const float*)d_in[7];
    const float* roll_wih   = (const float*)d_in[8];
    const float* roll_whh   = (const float*)d_in[9];
    const float* roll_bih   = (const float*)d_in[10];
    const float* roll_bhh   = (const float*)d_in[11];
    const float* unroll_wih = (const float*)d_in[12];
    const float* unroll_whh = (const float*)d_in[13];
    const float* unroll_bih = (const float*)d_in[14];
    const float* unroll_bhh = (const float*)d_in[15];
    const float* cls_w      = (const float*)d_in[16];
    const float* cls_b      = (const float*)d_in[17];

    char* ws = (char*)d_ws;
    size_t off = 0;
#define WSALLOC(T, name, count) \
    T* name = (T*)(ws + off);   \
    off += (((size_t)(count) * sizeof(T)) + 255) & ~(size_t)255;

    WSALLOC(bf16,  feats_bf,    (size_t)S_ * B_ * FEAT_)
    WSALLOC(bf16,  hands_bf,    (size_t)S_ * B_ * 128)
    WSALLOC(bf16,  x_bf,        (size_t)S_ * B_ * 2 * H_)
    WSALLOC(bf16,  fc1w_bf,     (size_t)H_ * FEAT_)
    WSALLOC(bf16,  fchw_bf,     (size_t)H_ * 128)
    WSALLOC(bf16,  rwih_bf,     (size_t)4 * H_ * 2 * H_)   // gate-permuted
    WSALLOC(bf16,  rwhh_bf,     (size_t)4 * H_ * H_)       // gate-permuted
    WSALLOC(bf16,  uwih_bf,     (size_t)4 * H_ * 2 * H_)   // gate-permuted
    WSALLOC(bf16,  uwhh_bf,     (size_t)4 * H_ * H_)       // gate-permuted
    WSALLOC(bf16,  clsw_bf,     (size_t)CPAD_ * H_)
    WSALLOC(float, roll_bsum,   4 * H_)                    // gate-permuted
    WSALLOC(float, unroll_bsum, 4 * H_)                    // gate-permuted
    WSALLOC(float, G,           (size_t)S_ * B_ * 4 * H_)  // permuted gates
    WSALLOC(float, Gu,          (size_t)B_ * 4 * H_)       // permuted gates
    WSALLOC(float, cvec,        (size_t)B_ * H_)
    WSALLOC(bf16,  h0_bf,       (size_t)B_ * H_)
    WSALLOC(bf16,  h1_bf,       (size_t)B_ * H_)
#undef WSALLOC

    const int NT = 256;
#define GRID1(n) dim3(((n) + NT - 1) / NT)

    // ---- pack / convert operands to bf16 (gate weights permuted p=4j+g) ----
    cvt_f32_bf16 <<<GRID1(S_*B_*FEAT_), NT, 0, stream>>>(feats,      feats_bf, S_*B_*FEAT_);
    pack_hands   <<<GRID1(S_*B_*128),   NT, 0, stream>>>(lh, rh,     hands_bf);
    cvt_f32_bf16 <<<GRID1(H_*FEAT_),    NT, 0, stream>>>(fc1_w,      fc1w_bf,  H_*FEAT_);
    cvt_fch      <<<GRID1(H_*128),      NT, 0, stream>>>(fch_w,      fchw_bf);
    cvt_w_perm   <<<GRID1(4*H_*2*H_),   NT, 0, stream>>>(roll_wih,   rwih_bf,  2*H_);
    cvt_w_perm   <<<GRID1(4*H_*H_),     NT, 0, stream>>>(roll_whh,   rwhh_bf,  H_);
    cvt_w_perm   <<<GRID1(4*H_*2*H_),   NT, 0, stream>>>(unroll_wih, uwih_bf,  2*H_);
    cvt_w_perm   <<<GRID1(4*H_*H_),     NT, 0, stream>>>(unroll_whh, uwhh_bf,  H_);
    cvt_cls      <<<GRID1(CPAD_*H_),    NT, 0, stream>>>(cls_w,      clsw_bf);
    bias_sum_perm<<<GRID1(4*H_),        NT, 0, stream>>>(roll_bih,   roll_bhh,   roll_bsum);
    bias_sum_perm<<<GRID1(4*H_),        NT, 0, stream>>>(unroll_bih, unroll_bhh, unroll_bsum);
    zero_hc      <<<GRID1(B_*H_),       NT, 0, stream>>>(cvec, h0_bf);

    // ---- x = [feats@fc1_w^T + b | hands@fch_w^T + b]  (bf16, S*B x 2H) ----
    wmma_gemm<4><<<dim3(H_/512, (S_*B_)/16), 256, 0, stream>>>(
        feats_bf, FEAT_, fc1w_bf, FEAT_, FEAT_,
        fc1_b, nullptr, 0, x_bf, 2*H_, H_);
    wmma_gemm<4><<<dim3(H_/512, (S_*B_)/16), 256, 0, stream>>>(
        hands_bf, 128, fchw_bf, 128, 128,
        fch_b, nullptr, 0, x_bf + H_, 2*H_, H_);

    // ---- precompute non-recurrent gate terms (permuted layout) ----
    wmma_gemm<4><<<dim3(4*H_/512, (S_*B_)/16), 256, 0, stream>>>(
        x_bf, 2*H_, rwih_bf, 2*H_, 2*H_,
        roll_bsum, G, 4*H_, nullptr, 0, 4*H_);
    wmma_gemm<4><<<dim3(4*H_/512, B_/16), 256, 0, stream>>>(
        x_bf + (size_t)(S_-1)*B_*2*H_, 2*H_, uwih_bf, 2*H_, 2*H_,
        unroll_bsum, Gu, 4*H_, nullptr, 0, 4*H_);

    // ---- sequential LSTM: one fused kernel per step, h ping-pong ----
    bf16* h_in  = h0_bf;
    bf16* h_out = h1_bf;
    for (int t = 0; t < S_; ++t) {
        lstm_step_fused<<<dim3(4*H_/256, B_/16), 256, 0, stream>>>(
            h_in, rwhh_bf, G + (size_t)t*B_*4*H_, cvec, h_out);
        bf16* tmp = h_in; h_in = h_out; h_out = tmp;
    }
    // only 2 unroll steps reach the output (h_n[S-1] = hseq[1])
    for (int u = 0; u < 2; ++u) {
        lstm_step_fused<<<dim3(4*H_/256, B_/16), 256, 0, stream>>>(
            h_in, uwhh_bf, Gu, cvec, h_out);
        bf16* tmp = h_in; h_in = h_out; h_out = tmp;
    }

    // ---- classifier: out = h @ cls_w^T + b  (rows padded 1000 -> 1024) ----
    float* out_f = (float*)d_out;
    wmma_gemm<2><<<dim3(CPAD_/256, B_/16), 256, 0, stream>>>(
        h_in, H_, clsw_bf, H_, H_,
        cls_b, out_f, C_, nullptr, 0, C_);

    // ---- pred (as float) + loss ----
    head_kernel<<<1, 128, 0, stream>>>(out_f, act,
                                       out_f + (size_t)B_*C_,
                                       out_f + (size_t)B_*C_ + B_);
#undef GRID1
}
